// TransformerBlock_69758858821804
// MI455X (gfx1250) — matrix operand order repaired and checked by reading, not compiled, not run on hardware
//
#include <hip/hip_runtime.h>
#include <hip/hip_bf16.h>
#include <math.h>

// ---------------------------------------------------------------------------
// Transformer block for MI455X (gfx1250, wave32, WMMA + TDM).
// N=4, T=2048, C=1024, H=16, DQK=DV=64, DH=4096.  ~260 GFLOP/call, compute
// bound -> GEMMs on v_wmma_f32_16x16x32_bf16; A operands staged to LDS by the
// Tensor Data Mover (double buffered), B register-pipelined.
// ---------------------------------------------------------------------------

typedef __bf16 bf16;
typedef unsigned int u32;
typedef __attribute__((ext_vector_type(16))) __bf16 v16bf;
typedef __attribute__((ext_vector_type(8)))  float  v8f;
typedef __attribute__((ext_vector_type(4)))  u32    u32x4;
typedef __attribute__((ext_vector_type(8)))  int    i32x8;
typedef __attribute__((ext_vector_type(4)))  int    i32x4;

union V16B { v16bf v; uint4 u[2]; };

#define TT   2048
#define CC   1024
#define HH   16
#define DHD  4096
#define NB   4
#define MM   (NB * TT)          // 8192 rows
#define KC   256                // K-chunk staged to LDS by TDM

__device__ __forceinline__ v8f wmma_bf16(v16bf a, v16bf b, v8f c) {
  // D = A(16x32 bf16) * B(32x16 bf16) + C(16x16 f32)
  return __builtin_amdgcn_wmma_f32_16x16x32_bf16(false, a, false, b,
                                                 (short)0, c, false, false);
}

// B operand = W^T where W is row-major [N][K] bf16.  Contiguous B k order used here
// (32x16): lane holds column n = lane&15; lanes 0-15 carry K k0..k0+15,
// lanes 16-31 carry K k0+16..k0+31 => one contiguous 32B chunk per lane.
__device__ __forceinline__ v16bf load_b(const bf16* W, int K, int n0, int k0,
                                        int lanelo, int lanehi) {
  const bf16* p = W + (size_t)(n0 + lanelo) * K + k0 + lanehi * 16;
  V16B r;
  r.u[0] = *reinterpret_cast<const uint4*>(p);
  r.u[1] = *reinterpret_cast<const uint4*>(p + 8);
  return r.v;
}

// A operand from an LDS-resident [16][KC] chunk.  Contiguous A k order used here (same as B):
// lane<16 -> M=lane, one contiguous 32B chunk at K k0+klo,
// klo = (lane>=16)?16:0.
__device__ __forceinline__ v16bf load_a_lds(const bf16* ab, int k0,
                                            int lanelo, int lanehi) {
  const bf16* p = ab + lanelo * KC + k0 + lanehi * 16;
  V16B r;
  r.u[0] = *reinterpret_cast<const uint4*>(p);
  r.u[1] = *reinterpret_cast<const uint4*>(p + 8);
  return r.v;
}

// ---------------------------------------------------------------------------
// Tensor Data Mover: DMA a 2D bf16 tile (tile_d1 rows x tile_d0 cols, row
// stride stride_d0 elements) from global memory into LDS at lds_addr.
// D# layout per CDNA5 ISA (group0 128b, group1 256b).
// ---------------------------------------------------------------------------
__device__ __forceinline__ void tdm_load_2d(u32 lds_addr, const bf16* gptr,
                                            u32 tile_d0, u32 tile_d1,
                                            u32 tensor_d0, u32 tensor_d1,
                                            u32 stride_d0) {
  unsigned long long ga = (unsigned long long)(const void*)gptr;
  u32x4 g0;
  g0[0] = 1u;                                            // count=1, user D#
  g0[1] = lds_addr;                                      // lds_addr [63:32]
  g0[2] = (u32)(ga & 0xffffffffu);                       // global_addr lo
  g0[3] = (u32)((ga >> 32) & 0x01ffffffu) | (2u << 30);  // addr[56:32] | type=2
  i32x8 g1;
  g1[0] = (int)(1u << 16);                               // data_size=1 (2B)
  g1[1] = (int)((tensor_d0 & 0xffffu) << 16);            // tensor_dim0 lo16
  g1[2] = (int)(((tensor_d0 >> 16) & 0xffffu) |
                ((tensor_d1 & 0xffffu) << 16));          // dim0 hi | dim1 lo
  g1[3] = (int)(((tensor_d1 >> 16) & 0xffffu) |
                ((tile_d0 & 0xffffu) << 16));            // dim1 hi | tile_dim0
  g1[4] = (int)(tile_d1 & 0xffffu);                      // tile_dim1 (dim2=0)
  g1[5] = (int)stride_d0;                                // tensor_dim0_stride lo
  g1[6] = 0;                                             // stride hi, dim1_stride
  g1[7] = 0;
  i32x4 z4 = {0, 0, 0, 0};
#if defined(__clang_major__) && __clang_major__ >= 23
  i32x8 z8 = {0, 0, 0, 0, 0, 0, 0, 0};
  __builtin_amdgcn_tensor_load_to_lds(g0, g1, z4, z4, z8, 0);
#else
  __builtin_amdgcn_tensor_load_to_lds(g0, g1, z4, z4, 0);
#endif
}

// ---------------------------------------------------------------------------
// Staged 16x64 GEMM tile: all 8 waves of the block share rows m0..m0+15.
// TDM double-buffers [16][KC] A-chunks in LDS; B is register-pipelined from
// global; 4 WMMAs per k-step reuse the A operand.
// ---------------------------------------------------------------------------
__device__ __forceinline__ void gemm_tile_16x64_staged(
    const bf16* __restrict__ A, const bf16* __restrict__ W, int K,
    int m0, int n0, int lanelo, int lanehi, bf16* abuf, v8f acc[4]) {
  const int nchunks = K / KC;
  if (threadIdx.x == 0) {
    tdm_load_2d((u32)(unsigned long long)abuf, A + (size_t)m0 * K,
                KC, 16, (u32)K, (u32)MM, (u32)K);
    __builtin_amdgcn_s_wait_tensorcnt(0);
  }
  __syncthreads();
  for (int kb = 0; kb < nchunks; ++kb) {
    const bf16* ab = abuf + (kb & 1) * (16 * KC);
    int kg = kb * KC;
    if (threadIdx.x == 0 && kb + 1 < nchunks) {
      tdm_load_2d((u32)(unsigned long long)(abuf + ((kb + 1) & 1) * (16 * KC)),
                  A + (size_t)m0 * K + kg + KC, KC, 16, (u32)K, (u32)MM, (u32)K);
    }
    if (kg + KC < K)   // prefetch B one chunk ahead (global_prefetch_b8)
      __builtin_prefetch(W + (size_t)(n0 + lanelo) * K + kg + KC, 0, 3);
    v16bf bcur[4];
#pragma unroll
    for (int j = 0; j < 4; ++j)
      bcur[j] = load_b(W, K, n0 + j * 16, kg, lanelo, lanehi);
#pragma unroll
    for (int k0 = 0; k0 < KC; k0 += 32) {
      v16bf a = load_a_lds(ab, k0, lanelo, lanehi);
      v16bf bnxt[4];
      if (k0 + 32 < KC) {
#pragma unroll
        for (int j = 0; j < 4; ++j)
          bnxt[j] = load_b(W, K, n0 + j * 16, kg + k0 + 32, lanelo, lanehi);
      }
#pragma unroll
      for (int j = 0; j < 4; ++j) acc[j] = wmma_bf16(a, bcur[j], acc[j]);
#pragma unroll
      for (int j = 0; j < 4; ++j) bcur[j] = bnxt[j];
    }
    if (threadIdx.x == 0 && kb + 1 < nchunks)
      __builtin_amdgcn_s_wait_tensorcnt(0);
    __syncthreads();
  }
}

__device__ __forceinline__ float gelu_f(float v) {
  const float c = 0.7978845608028654f;
  float t = tanhf(c * (v + 0.044715f * v * v * v));
  return 0.5f * v * (1.0f + t);
}

// ---------------------------------------------------------------------------
// fp32 -> bf16 weight conversion
// ---------------------------------------------------------------------------
__global__ void __launch_bounds__(256) cvt_bf16_kernel(const float* __restrict__ src,
                                                       bf16* __restrict__ dst, int n) {
  int i = blockIdx.x * 256 + threadIdx.x;
  if (i < n) dst[i] = (bf16)src[i];
}

// ---------------------------------------------------------------------------
// LayerNorm over C=1024, fp32 in, bf16 out (GEMM A operand)
// ---------------------------------------------------------------------------
__global__ void __launch_bounds__(256) ln_kernel(const float* __restrict__ x,
                                                 const float* __restrict__ w,
                                                 const float* __restrict__ b,
                                                 bf16* __restrict__ out) {
  __shared__ float red[256];
  int r = blockIdx.x, tid = threadIdx.x;
  const float4 f = reinterpret_cast<const float4*>(x + (size_t)r * CC)[tid];
  red[tid] = f.x + f.y + f.z + f.w;
  __syncthreads();
  for (int o2 = 128; o2 > 0; o2 >>= 1) {
    if (tid < o2) red[tid] += red[tid + o2];
    __syncthreads();
  }
  float mu = red[0] * (1.0f / CC);
  __syncthreads();
  float dx = f.x - mu, dy = f.y - mu, dz = f.z - mu, dw = f.w - mu;
  red[tid] = dx * dx + dy * dy + dz * dz + dw * dw;
  __syncthreads();
  for (int o2 = 128; o2 > 0; o2 >>= 1) {
    if (tid < o2) red[tid] += red[tid + o2];
    __syncthreads();
  }
  float rstd = rsqrtf(red[0] * (1.0f / CC) + 1e-5f);
  const float4 wv = reinterpret_cast<const float4*>(w)[tid];
  const float4 bv = reinterpret_cast<const float4*>(b)[tid];
  bf16* orow = out + (size_t)r * CC + tid * 4;
  orow[0] = (bf16)(dx * rstd * wv.x + bv.x);
  orow[1] = (bf16)(dy * rstd * wv.y + bv.y);
  orow[2] = (bf16)(dz * rstd * wv.z + bv.z);
  orow[3] = (bf16)(dw * rstd * wv.w + bv.w);
}

// ---------------------------------------------------------------------------
// QKV projection: [8192,1024] x [3072,1024]^T, scatter into attention layouts
//   q,k: [N][H][T][64] bf16     v: transposed [N][H][64][T] bf16
// ---------------------------------------------------------------------------
__global__ void __launch_bounds__(256) gemm_qkv_kernel(const bf16* __restrict__ xb,
                                                       const bf16* __restrict__ wqkv,
                                                       bf16* __restrict__ qb,
                                                       bf16* __restrict__ kb,
                                                       bf16* __restrict__ vt) {
  __shared__ __align__(16) bf16 abuf[2][16][KC];
  int wid = (blockIdx.x * 256 + threadIdx.x) >> 5;
  int lane = threadIdx.x & 31, lanelo = lane & 15, lanehi = lane >> 4;
  const int NT = 3072 / 64;
  int m0 = (wid / NT) * 16, n0 = (wid % NT) * 64;
  v8f acc[4] = {};
  gemm_tile_16x64_staged(xb, wqkv, CC, m0, n0, lanelo, lanehi,
                         &abuf[0][0][0], acc);
#pragma unroll
  for (int j = 0; j < 4; ++j) {
    int col = n0 + j * 16 + lanelo;         // 0..3071
    int proj = col >> 10, rem = col & 1023;
    int h = rem >> 6, d = rem & 63;
#pragma unroll
    for (int v = 0; v < 8; ++v) {
      int m = m0 + v + lanehi * 8;          // C-layout row
      int nbat = m >> 11, t = m & (TT - 1);
      bf16 val = (bf16)acc[j][v];
      size_t nh = (size_t)nbat * HH + h;
      if (proj == 0)      qb[(nh * TT + t) * 64 + d] = val;
      else if (proj == 1) kb[(nh * TT + t) * 64 + d] = val;
      else                vt[(nh * 64 + d) * TT + t] = val;
    }
  }
}

// ---------------------------------------------------------------------------
// Flash attention: one wave handles 16 q rows of one (n,h); causal, online
// softmax; P re-packed C-layout -> A-layout through per-wave LDS tile.
// Residual add fused: x1 = x + attn  (written to d_out).
// ---------------------------------------------------------------------------
__global__ void __launch_bounds__(256) attn_kernel(const bf16* __restrict__ qb,
                                                   const bf16* __restrict__ kb,
                                                   const bf16* __restrict__ vt,
                                                   const float* __restrict__ x,
                                                   float* __restrict__ x1) {
  __shared__ __align__(16) bf16 pb[8][16][32];   // 8 KB: per-wave P transpose tile
  int gw = (blockIdx.x * 256 + threadIdx.x) >> 5;
  int lane = threadIdx.x & 31, lanelo = lane & 15, lanehi = lane >> 4;
  int wslot = threadIdx.x >> 5;
  const int QT = TT / 16;
  int q0 = (gw % QT) * 16;
  int h = (gw / QT) % HH;
  int n = gw / (QT * HH);

  size_t nh = (size_t)n * HH + h;
  const bf16* Qp = qb + nh * TT * 64;
  const bf16* Kp = kb + nh * TT * 64;
  const bf16* Vp = vt + nh * 64 * TT;

  // Q tile as two A operands (d 0..31 and 32..63)
  v16bf qa[2];
  {
    const bf16* qrow = Qp + (size_t)(q0 + lanelo) * 64;
#pragma unroll
    for (int ds = 0; ds < 2; ++ds) {
      const bf16* p = qrow + ds * 32 + lanehi * 16;
      V16B r;
      r.u[0] = *reinterpret_cast<const uint4*>(p);
      r.u[1] = *reinterpret_cast<const uint4*>(p + 8);
      qa[ds] = r.v;
    }
  }

  v8f o[4] = {};
  float mrow[8], lrow[8];
#pragma unroll
  for (int v = 0; v < 8; ++v) { mrow[v] = -3.0e38f; lrow[v] = 0.0f; }

  for (int s0 = 0; s0 <= q0 + 15; s0 += 32) {
    float p0[8], p1[8];
    // ---- S = (Q K^T) * 1/sqrt(64), two 16-col groups, causal mask ----
#pragma unroll
    for (int sc = 0; sc < 2; ++sc) {
      v8f sacc = {};
      int scol = s0 + sc * 16 + lanelo;
      int scl = scol < TT - 1 ? scol : TT - 1;   // clamp OOB (masked below)
#pragma unroll
      for (int ds = 0; ds < 2; ++ds) {
        const bf16* p = Kp + (size_t)scl * 64 + ds * 32 + lanehi * 16;
        V16B r;
        r.u[0] = *reinterpret_cast<const uint4*>(p);
        r.u[1] = *reinterpret_cast<const uint4*>(p + 8);
        sacc = wmma_bf16(qa[ds], r.v, sacc);
      }
#pragma unroll
      for (int v = 0; v < 8; ++v) {
        int rowg = q0 + v + lanehi * 8;
        float e = sacc[v] * 0.125f;
        e = (scol > rowg) ? -3.0e38f : e;
        if (sc == 0) p0[v] = e; else p1[v] = e;
      }
    }
    // ---- online softmax: row reductions across the 16 lanes of each half ----
    float mx[8];
#pragma unroll
    for (int v = 0; v < 8; ++v) mx[v] = fmaxf(p0[v], p1[v]);
#pragma unroll
    for (int msk = 1; msk < 16; msk <<= 1)
#pragma unroll
      for (int v = 0; v < 8; ++v)
        mx[v] = fmaxf(mx[v], __shfl_xor(mx[v], msk, 32));
    float alpha[8];
#pragma unroll
    for (int v = 0; v < 8; ++v) {
      float mnew = fmaxf(mrow[v], mx[v]);
      alpha[v] = __expf(mrow[v] - mnew);
      p0[v] = __expf(p0[v] - mnew);
      p1[v] = __expf(p1[v] - mnew);
      mrow[v] = mnew;
    }
    float rs[8];
#pragma unroll
    for (int v = 0; v < 8; ++v) rs[v] = p0[v] + p1[v];
#pragma unroll
    for (int msk = 1; msk < 16; msk <<= 1)
#pragma unroll
      for (int v = 0; v < 8; ++v)
        rs[v] += __shfl_xor(rs[v], msk, 32);
#pragma unroll
    for (int v = 0; v < 8; ++v) lrow[v] = lrow[v] * alpha[v] + rs[v];
#pragma unroll
    for (int j = 0; j < 4; ++j)
#pragma unroll
      for (int v = 0; v < 8; ++v) o[j][v] *= alpha[v];

    // ---- P: C-layout -> A-layout via wave-private LDS tile ----
#pragma unroll
    for (int v = 0; v < 8; ++v) {
      int row = v + lanehi * 8;
      pb[wslot][row][lanelo]      = (bf16)p0[v];
      pb[wslot][row][lanelo + 16] = (bf16)p1[v];
    }
    asm volatile("s_wait_dscnt 0x0" ::: "memory");  // wave-local DS RAW fence
    v16bf pa;
    {
      const bf16* p = &pb[wslot][lanelo][lanehi * 16];
      V16B r;
      r.u[0] = *reinterpret_cast<const uint4*>(p);
      r.u[1] = *reinterpret_cast<const uint4*>(p + 8);
      pa = r.v;
    }
    asm volatile("" ::: "memory");

    // ---- O += P * V  (V stored transposed: contiguous along s) ----
#pragma unroll
    for (int j = 0; j < 4; ++j) {
      int dcol = j * 16 + lanelo;
      int sb = s0 + lanehi * 16;
      if (sb > TT - 16) sb = TT - 16;       // clamp; clamped cols have P==0
      const bf16* p = Vp + (size_t)dcol * TT + sb;
      V16B r;
      r.u[0] = *reinterpret_cast<const uint4*>(p);
      r.u[1] = *reinterpret_cast<const uint4*>(p + 8);
      o[j] = wmma_bf16(pa, r.v, o[j]);
    }
  }

  // ---- epilogue: y = O / l ; x1 = x + y ----
#pragma unroll
  for (int v = 0; v < 8; ++v) {
    float inv = 1.0f / lrow[v];
    int t = q0 + v + lanehi * 8;
#pragma unroll
    for (int j = 0; j < 4; ++j) {
      int c = h * 64 + j * 16 + lanelo;
      size_t idx = ((size_t)n * TT + t) * CC + c;
      x1[idx] = x[idx] + o[j][v] * inv;
    }
  }
}

// ---------------------------------------------------------------------------
// fc1: [8192,1024] x [4096,1024]^T, bias + GELU, bf16 out
// ---------------------------------------------------------------------------
__global__ void __launch_bounds__(256) gemm_fc1_kernel(const bf16* __restrict__ xb,
                                                       const bf16* __restrict__ w,
                                                       const float* __restrict__ bias,
                                                       bf16* __restrict__ act) {
  __shared__ __align__(16) bf16 abuf[2][16][KC];
  int wid = (blockIdx.x * 256 + threadIdx.x) >> 5;
  int lane = threadIdx.x & 31, lanelo = lane & 15, lanehi = lane >> 4;
  const int NT = DHD / 64;
  int m0 = (wid / NT) * 16, n0 = (wid % NT) * 64;
  v8f acc[4] = {};
  gemm_tile_16x64_staged(xb, w, CC, m0, n0, lanelo, lanehi, &abuf[0][0][0], acc);
#pragma unroll
  for (int j = 0; j < 4; ++j) {
    int col = n0 + j * 16 + lanelo;
    float bz = bias[col];
#pragma unroll
    for (int v = 0; v < 8; ++v) {
      int m = m0 + v + lanehi * 8;
      act[(size_t)m * DHD + col] = (bf16)gelu_f(acc[j][v] + bz);
    }
  }
}

// ---------------------------------------------------------------------------
// fc2: [8192,4096] x [1024,4096]^T, bias + residual (x1 lives in d_out)
// ---------------------------------------------------------------------------
__global__ void __launch_bounds__(256) gemm_fc2_kernel(const bf16* __restrict__ act,
                                                       const bf16* __restrict__ w,
                                                       const float* __restrict__ bias,
                                                       float* __restrict__ out) {
  __shared__ __align__(16) bf16 abuf[2][16][KC];
  int wid = (blockIdx.x * 256 + threadIdx.x) >> 5;
  int lane = threadIdx.x & 31, lanelo = lane & 15, lanehi = lane >> 4;
  const int NT = CC / 64;
  int m0 = (wid / NT) * 16, n0 = (wid % NT) * 64;
  v8f acc[4] = {};
  gemm_tile_16x64_staged(act, w, DHD, m0, n0, lanelo, lanehi, &abuf[0][0][0], acc);
#pragma unroll
  for (int j = 0; j < 4; ++j) {
    int col = n0 + j * 16 + lanelo;
    float bz = bias[col];
#pragma unroll
    for (int v = 0; v < 8; ++v) {
      int m = m0 + v + lanehi * 8;
      size_t idx = (size_t)m * CC + col;
      out[idx] = out[idx] + acc[j][v] + bz;   // out held x1 (residual stream)
    }
  }
}

// ---------------------------------------------------------------------------
// Host: workspace carving + launch sequence (all on `stream`, graph-safe).
// Workspace requirement ~157 MB.
// ---------------------------------------------------------------------------
extern "C" void kernel_launch(void* const* d_in, const int* in_sizes, int n_in,
                              void* d_out, int out_size, void* d_ws, size_t ws_size,
                              hipStream_t stream) {
  const float* x     = (const float*)d_in[0];
  const float* ln1_w = (const float*)d_in[1];
  const float* ln1_b = (const float*)d_in[2];
  const float* w_q   = (const float*)d_in[3];
  const float* w_k   = (const float*)d_in[4];
  const float* w_v   = (const float*)d_in[5];
  const float* ln2_w = (const float*)d_in[6];
  const float* ln2_b = (const float*)d_in[7];
  const float* fc1_w = (const float*)d_in[8];
  const float* fc1_b = (const float*)d_in[9];
  const float* fc2_w = (const float*)d_in[10];
  const float* fc2_b = (const float*)d_in[11];
  float* out = (float*)d_out;

  char* wsb = (char*)d_ws;
  size_t off = 0;
  auto carve = [&](size_t bytes) -> void* {
    void* p = wsb + off;
    off = (off + bytes + 255) & ~(size_t)255;
    return p;
  };
  bf16* xb    = (bf16*)carve((size_t)MM * CC * 2);         // LN output (reused ln1/ln2)
  bf16* wqkv  = (bf16*)carve((size_t)3072 * CC * 2);       // packed [wq;wk;wv]
  bf16* fc1wb = (bf16*)carve((size_t)DHD * CC * 2);
  bf16* fc2wb = (bf16*)carve((size_t)CC * DHD * 2);
  bf16* qb    = (bf16*)carve((size_t)NB * HH * TT * 64 * 2);
  bf16* kb    = (bf16*)carve((size_t)NB * HH * TT * 64 * 2);
  bf16* vt    = (bf16*)carve((size_t)NB * HH * 64 * TT * 2);
  bf16* act   = (bf16*)carve((size_t)MM * DHD * 2);

  const int WPROJ = HH * 64 * CC;      // 1,048,576 per projection
  const int WFC   = DHD * CC;          // 4,194,304
  cvt_bf16_kernel<<<(WPROJ + 255) / 256, 256, 0, stream>>>(w_q, wqkv, WPROJ);
  cvt_bf16_kernel<<<(WPROJ + 255) / 256, 256, 0, stream>>>(w_k, wqkv + WPROJ, WPROJ);
  cvt_bf16_kernel<<<(WPROJ + 255) / 256, 256, 0, stream>>>(w_v, wqkv + 2 * WPROJ, WPROJ);
  cvt_bf16_kernel<<<(WFC + 255) / 256, 256, 0, stream>>>(fc1_w, fc1wb, WFC);
  cvt_bf16_kernel<<<(WFC + 255) / 256, 256, 0, stream>>>(fc2_w, fc2wb, WFC);

  // ln1(x) -> xb
  ln_kernel<<<MM, 256, 0, stream>>>(x, ln1_w, ln1_b, xb);
  // qkv projection (24576 waves / 8 per block)
  gemm_qkv_kernel<<<(MM / 16) * (3072 / 64) / 8, 256, 0, stream>>>(xb, wqkv, qb, kb, vt);
  // flash attention + residual -> out (= x1)
  attn_kernel<<<NB * HH * (TT / 16) / 8, 256, 0, stream>>>(qb, kb, vt, x, out);
  // ln2(x1) -> xb
  ln_kernel<<<MM, 256, 0, stream>>>(out, ln2_w, ln2_b, xb);
  // fc1 + GELU -> act
  gemm_fc1_kernel<<<(MM / 16) * (DHD / 64) / 8, 256, 0, stream>>>(xb, fc1wb, fc1_b, act);
  // fc2 + bias + residual -> out
  gemm_fc2_kernel<<<(MM / 16) * (CC / 64) / 8, 256, 0, stream>>>(act, fc2wb, fc2_b, out);
  (void)n_in; (void)in_sizes; (void)out_size; (void)ws_size;
}
